// ViLLayer_8100308321067
// MI455X (gfx1250) — matrix operand order repaired and checked by reading, not compiled, not run on hardware
//
#include <hip/hip_runtime.h>
#include <hip/hip_bf16.h>

// ---------------- problem constants (from reference) ----------------
#define BATCH   2
#define DIM     96
#define INNER   192        // DIM * EXPANSION
#define NH      48
#define DH      4
#define CONV_K  4
#define S_LEN   576        // H*W = 24*24
#define BS      (BATCH * S_LEN)   // 1152
#define UP_N    (2 * INNER)       // 384
#define QKV_F   (3 * INNER)       // 576

typedef __attribute__((ext_vector_type(16))) __bf16 v16bf;
typedef __attribute__((ext_vector_type(8)))  __bf16 v8bf;
typedef __attribute__((ext_vector_type(8)))  float  v8f;

// ---- TDM availability (device pass only; arity differs per toolchain) ----
#if defined(__gfx1250__) && __has_builtin(__builtin_amdgcn_tensor_load_to_lds)
#define USE_TDM 1
#else
#define USE_TDM 0
#endif

#if USE_TDM
typedef unsigned int u32;
typedef __attribute__((ext_vector_type(4))) u32 v4u;
typedef __attribute__((ext_vector_type(8))) int v8i_t;
typedef __attribute__((ext_vector_type(4))) int v4i_t;

// Build a 2D D# (group0/group1 per CDNA5 ISA) and issue
// TENSOR_LOAD_TO_LDS. dims/strides in 4-byte elements (data_size=4B).
__device__ __forceinline__ void tdm_load_2d(const void* gptr, void* lptr,
                                            u32 tensor_d0, u32 tensor_d1,
                                            u32 tile_d0, u32 tile_d1, u32 stride0) {
  const unsigned long long ga = (unsigned long long)gptr;
  const u32 lds = (u32)(unsigned long long)lptr;
  v4u g0;
  g0[0] = 1u;                                               // count=1, user descriptor
  g0[1] = lds;                                              // lds_addr [63:32]
  g0[2] = (u32)ga;                                          // global_addr [95:64]
  g0[3] = (u32)((ga >> 32) & 0x1FFFFFFull) | (2u << 30);    // addr[56:32] | type=2
  v8i_t g1;
  g1[0] = (int)(2u << 16);                                  // wg_mask=0, data_size=2 (4B)
  g1[1] = (int)((tensor_d0 & 0xffffu) << 16);               // tensor_dim0[15:0] @ [63:48]
  g1[2] = (int)(((tensor_d0 >> 16) & 0xffffu) |             // tensor_dim0[31:16]
                ((tensor_d1 & 0xffffu) << 16));             // tensor_dim1[15:0]
  g1[3] = (int)(((tensor_d1 >> 16) & 0xffffu) |             // tensor_dim1[31:16]
                ((tile_d0 & 0xffffu) << 16));               // tile_dim0
  g1[4] = (int)(tile_d1 & 0xffffu);                         // tile_dim1 (tile_dim2=0)
  g1[5] = (int)stride0;                                     // tensor_dim0_stride[31:0]
  g1[6] = 0;                                                // stride hi / dim1_stride
  g1[7] = 0;
  v4i_t z4 = {};
#if __has_include(<hip/amd_detail/amd_gfx1250_TDM.h>)
  v8i_t z8 = {};
  __builtin_amdgcn_tensor_load_to_lds(g0, g1, z4, z4, z8, 0);   // clang-23 6-arg form
#else
  __builtin_amdgcn_tensor_load_to_lds(g0, g1, z4, z4, 0);       // ROCm 7.2 5-arg form
#endif
}
#endif  // USE_TDM

// ---- WMMA fragment loaders (bf16 16x16x32, contiguous k = 16*half + s in both A and B) ----
// A: 16x32 (MxK) bf16 row-major. Per lane: rows = lane&15, two contiguous
// 8-element chunks at K = k0 + (lane>=16 ? 16 : 0) and +8.
__device__ __forceinline__ v16bf load_a_frag_bf(const __bf16* __restrict__ A, int lda,
                                                int m0, int k0, int lane) {
  const int r = lane & 15, half = lane >> 4;
  const __bf16* row = A + (size_t)(m0 + r) * lda + k0 + half * 16;
  union { v16bf v; v8bf h[2]; } u;
  u.h[0] = *(const v8bf*)(row);
  u.h[1] = *(const v8bf*)(row + 8);
  return u.v;
}

// B: 32x16 (KxN) from weight row W[n][.]: per lane one contiguous 16-element
// chunk at K = k0 + (lane>=16 ? 16 : 0).
__device__ __forceinline__ v16bf load_b_frag_bf(const __bf16* __restrict__ wrow,
                                                int k0, int lane) {
  const int half = lane >> 4;
  return *(const v16bf*)(wrow + k0 + half * 16);
}

// ---------------- K0: f32 -> bf16 weight staging ----------------
__global__ __launch_bounds__(256) void k_cvt_bf16(const float* __restrict__ s,
                                                  __bf16* __restrict__ d, int n) {
  const int i = blockIdx.x * blockDim.x + threadIdx.x;
  if (i < n) d[i] = (__bf16)s[i];
}

// ---------------- K1: LayerNorm (one wave32 per (b,s)), bf16 out ----------------
__global__ __launch_bounds__(32) void k_layernorm(const float* __restrict__ x,
                                                  const float* __restrict__ g,
                                                  const float* __restrict__ be,
                                                  __bf16* __restrict__ hout) {
  const int bs = blockIdx.x;
  const int b = bs / S_LEN, s = bs % S_LEN;
  const int lane = threadIdx.x;
  float vals[3], sum = 0.f, sq = 0.f;
#pragma unroll
  for (int i = 0; i < 3; ++i) {
    const int c = lane + 32 * i;
    const float v = x[((size_t)b * DIM + c) * S_LEN + s];
    vals[i] = v; sum += v; sq += v * v;
  }
#pragma unroll
  for (int off = 16; off; off >>= 1) {
    sum += __shfl_xor(sum, off, 32);
    sq  += __shfl_xor(sq,  off, 32);
  }
  const float mean = sum * (1.f / DIM);
  const float var  = sq * (1.f / DIM) - mean * mean;
  const float inv  = rsqrtf(var + 1e-5f);
#pragma unroll
  for (int i = 0; i < 3; ++i) {
    const int c = lane + 32 * i;
    hout[(size_t)bs * DIM + c] = (__bf16)((vals[i] - mean) * inv * g[c] + be[c]);
  }
}

// ---------------- K2: up-projection GEMM (WMMA) ----------------
__global__ __launch_bounds__(32) void k_up_gemm(const __bf16* __restrict__ A,   // (1152,96)
                                                const __bf16* __restrict__ W,   // (384,96)
                                                const float* __restrict__ bias,
                                                float* __restrict__ out) {      // (1152,384)
  const int nt = blockIdx.x % (UP_N / 16), mt = blockIdx.x / (UP_N / 16);
  const int m0 = mt * 16, n0 = nt * 16;
  const int lane = threadIdx.x;
  const int n = n0 + (lane & 15);
  const __bf16* wrow = W + (size_t)n * DIM;
  v8f c = {};
#pragma unroll
  for (int k0 = 0; k0 < DIM; k0 += 32) {
    v16bf a = load_a_frag_bf(A, DIM, m0, k0, lane);
    v16bf b = load_b_frag_bf(wrow, k0, lane);
    c = __builtin_amdgcn_wmma_f32_16x16x32_bf16(false, a, false, b, (short)0, c, false, false);
  }
  const int half = lane >> 4;
  const float bn = bias[n];
#pragma unroll
  for (int i = 0; i < 8; ++i) {
    const int m = m0 + i + half * 8;
    out[(size_t)m * UP_N + n] = c[i] + bn;
  }
}

// ---------------- K3: causal depthwise conv + SiLU ----------------
__global__ __launch_bounds__(256) void k_conv_silu(const float* __restrict__ xi,
                                                   const float* __restrict__ wconv,
                                                   const float* __restrict__ bconv,
                                                   float* __restrict__ xconv) {
  const int idx = blockIdx.x * blockDim.x + threadIdx.x;
  if (idx >= BS * INNER) return;
  const int c = idx % INNER;
  const int bs = idx / INNER;
  const int s = bs % S_LEN, b = bs / S_LEN;
  float acc = bconv[c];
#pragma unroll
  for (int kk = 0; kk < CONV_K; ++kk) {
    const int sp = s - (CONV_K - 1) + kk;
    if (sp >= 0)
      acc += xi[(size_t)(b * S_LEN + sp) * UP_N + c] * wconv[c * CONV_K + kk];
  }
  xconv[idx] = acc / (1.f + __expf(-acc));       // silu
}

// ---------------- K4: head-wise 4x4 q/k/v projections (f32 + bf16 copies) ----------------
__global__ __launch_bounds__(256) void k_qkv(const float* __restrict__ xi,
                                             const float* __restrict__ xconv,
                                             const float* __restrict__ wq, const float* __restrict__ bq,
                                             const float* __restrict__ wk, const float* __restrict__ bk,
                                             const float* __restrict__ wv, const float* __restrict__ bv,
                                             float* __restrict__ qkv,
                                             __bf16* __restrict__ qkv_bf) {
  const int idx = blockIdx.x * blockDim.x + threadIdx.x;
  if (idx >= BS * INNER) return;
  const int ci = idx % INNER;
  const int bs = idx / INNER;
  const int h = ci >> 2, o = ci & 3;
  const float* xc = xconv + (size_t)bs * INNER + h * 4;
  const float* xm = xi + (size_t)bs * UP_N + h * 4;   // x_mlstm half of xi
  float q = bq[ci], k = bk[ci], v = bv[ci];
#pragma unroll
  for (int d = 0; d < 4; ++d) {
    const int w = (h * 4 + o) * 4 + d;
    q += xc[d] * wq[w];
    k += xc[d] * wk[w];
    v += xm[d] * wv[w];
  }
  float* row = qkv + (size_t)bs * QKV_F;
  row[ci] = q; row[INNER + ci] = k; row[2 * INNER + ci] = v;
  __bf16* rbf = qkv_bf + (size_t)bs * QKV_F;
  rbf[ci] = (__bf16)q; rbf[INNER + ci] = (__bf16)k; rbf[2 * INNER + ci] = (__bf16)v;
}

// ---------------- K5: gate GEMM (WMMA): ig/fg = qkv @ w^T ----------------
__global__ __launch_bounds__(32) void k_gate_gemm(const __bf16* __restrict__ qkv_bf,
                                                  const __bf16* __restrict__ w_ig,
                                                  const float* __restrict__ b_ig,
                                                  const __bf16* __restrict__ w_fg,
                                                  const float* __restrict__ b_fg,
                                                  float* __restrict__ ig,
                                                  float* __restrict__ fg) {
  const int nt = blockIdx.x % 6, mt = blockIdx.x / 6;   // N = 96 = 2*NH
  const int m0 = mt * 16, n0 = nt * 16;
  const int lane = threadIdx.x;
  const int n = n0 + (lane & 15);
  const bool is_fg = (n >= NH);
  const int hh = is_fg ? n - NH : n;
  const __bf16* wrow = (is_fg ? w_fg : w_ig) + (size_t)hh * QKV_F;
  v8f c = {};
  for (int k0 = 0; k0 < QKV_F; k0 += 32) {
    v16bf a = load_a_frag_bf(qkv_bf, QKV_F, m0, k0, lane);
    v16bf b = load_b_frag_bf(wrow, k0, lane);
    c = __builtin_amdgcn_wmma_f32_16x16x32_bf16(false, a, false, b, (short)0, c, false, false);
  }
  const int half = lane >> 4;
  const float bn = (is_fg ? b_fg : b_ig)[hh];
  float* dst = is_fg ? fg : ig;
#pragma unroll
  for (int i = 0; i < 8; ++i) {
    const int m = m0 + i + half * 8;
    const int b = m / S_LEN, s = m % S_LEN;
    dst[((size_t)b * NH + hh) * S_LEN + s] = c[i] + bn;
  }
}

// ---------------- K6: stabilized parallel mLSTM (TDM-staged LDS, online max) ----------------
__global__ __launch_bounds__(256) void k_mlstm(const float* __restrict__ qkv,
                                               const float* __restrict__ igb,
                                               const float* __restrict__ fgb,
                                               float* __restrict__ htil) {
  const int bh = blockIdx.x;
  const int b = bh / NH, h = bh % NH;
  __shared__ float4 sQ[S_LEN], sK[S_LEN], sV[S_LEN];
  __shared__ float  sC[S_LEN];     // fg -> logsigmoid -> inclusive cumsum
  __shared__ float  sI[S_LEN];     // ig
  const int tid = threadIdx.x;

#if USE_TDM
  if (tid < 32) {
    const float* qbase = qkv + (size_t)(b * S_LEN) * QKV_F + h * 4;
    // q/k/v: 576 rows x 4 elements, row stride 576 elements -> contiguous float4[576]
    tdm_load_2d(qbase,             sQ, QKV_F, S_LEN, 4, S_LEN, QKV_F);
    tdm_load_2d(qbase + INNER,     sK, QKV_F, S_LEN, 4, S_LEN, QKV_F);
    tdm_load_2d(qbase + 2 * INNER, sV, QKV_F, S_LEN, 4, S_LEN, QKV_F);
    // ig/fg: one contiguous 576-element row
    tdm_load_2d(igb + ((size_t)b * NH + h) * S_LEN, sI, S_LEN, 1, S_LEN, 1, S_LEN);
    tdm_load_2d(fgb + ((size_t)b * NH + h) * S_LEN, sC, S_LEN, 1, S_LEN, 1, S_LEN);
    __builtin_amdgcn_s_wait_tensorcnt(0);
  }
  __syncthreads();
  for (int t = tid; t < S_LEN; t += blockDim.x) {
    const float f = sC[t];
    sC[t] = fminf(f, 0.f) - log1pf(__expf(-fabsf(f)));   // log_sigmoid
  }
#else
  for (int t = tid; t < S_LEN; t += blockDim.x) {
    const float* row = qkv + (size_t)(b * S_LEN + t) * QKV_F;
    const int off = h * 4;
    sQ[t] = make_float4(row[off], row[off + 1], row[off + 2], row[off + 3]);
    sK[t] = make_float4(row[INNER + off], row[INNER + off + 1],
                        row[INNER + off + 2], row[INNER + off + 3]);
    sV[t] = make_float4(row[2 * INNER + off], row[2 * INNER + off + 1],
                        row[2 * INNER + off + 2], row[2 * INNER + off + 3]);
    sI[t] = igb[((size_t)b * NH + h) * S_LEN + t];
    const float f = fgb[((size_t)b * NH + h) * S_LEN + t];
    sC[t] = fminf(f, 0.f) - log1pf(__expf(-fabsf(f)));
  }
#endif
  __syncthreads();
  if (tid == 0) {               // sequential inclusive scan (S=576, cheap)
    float run = 0.f;
    for (int t = 0; t < S_LEN; ++t) { run += sC[t]; sC[t] = run; }
  }
  __syncthreads();

  const int wave = tid >> 5, lane = tid & 31;
  for (int s = wave; s < S_LEN; s += 8) {
    const float Cs = sC[s];
    const float4 qv = sQ[s];
    float m = -INFINITY, sum = 0.f, a0 = 0.f, a1 = 0.f, a2 = 0.f, a3 = 0.f;
    for (int t = lane; t <= s; t += 32) {
      const float ld = Cs - sC[t] + sI[t];                 // log_D[s,t]
      const float4 kv = sK[t];
      const float qk = 0.5f * (qv.x * kv.x + qv.y * kv.y + qv.z * kv.z + qv.w * kv.w);
      const float mn = fmaxf(m, ld);
      const float corr = (m == -INFINITY) ? 0.f : __expf(m - mn);
      const float p = __expf(ld - mn) * qk;                // Cm element (rescaled)
      const float4 vv = sV[t];
      sum = sum * corr + p;
      a0 = a0 * corr + p * vv.x;  a1 = a1 * corr + p * vv.y;
      a2 = a2 * corr + p * vv.z;  a3 = a3 * corr + p * vv.w;
      m = mn;
    }
    // butterfly combine across the wave (online-softmax merge)
#pragma unroll
    for (int off = 16; off; off >>= 1) {
      const float mo = __shfl_xor(m,   off, 32);
      const float so = __shfl_xor(sum, off, 32);
      const float b0 = __shfl_xor(a0,  off, 32);
      const float b1 = __shfl_xor(a1,  off, 32);
      const float b2 = __shfl_xor(a2,  off, 32);
      const float b3 = __shfl_xor(a3,  off, 32);
      const float mn = fmaxf(m, mo);
      const float c1 = (m  == -INFINITY) ? 0.f : __expf(m  - mn);
      const float c2 = (mo == -INFINITY) ? 0.f : __expf(mo - mn);
      sum = sum * c1 + so * c2;
      a0 = a0 * c1 + b0 * c2;  a1 = a1 * c1 + b1 * c2;
      a2 = a2 * c1 + b2 * c2;  a3 = a3 * c1 + b3 * c2;
      m = mn;
    }
    if (lane == 0) {
      const float norm = fmaxf(fabsf(sum), __expf(-m)) + 1e-6f;
      const float inv = 1.f / norm;
      ((float4*)htil)[((size_t)b * NH + h) * S_LEN + s] =
          make_float4(a0 * inv, a1 * inv, a2 * inv, a3 * inv);
    }
  }
}

// ---------------- K7: per-head norm + skip + SiLU gate (bf16 out) ----------------
__global__ __launch_bounds__(256) void k_fuse(const float* __restrict__ htil,
                                              const float* __restrict__ on_g,
                                              const float* __restrict__ on_b,
                                              const float* __restrict__ skip,
                                              const float* __restrict__ xconv,
                                              const float* __restrict__ xi,
                                              __bf16* __restrict__ hstate_bf) {
  const int idx = blockIdx.x * blockDim.x + threadIdx.x;
  if (idx >= BS * NH) return;
  const int h = idx % NH;
  const int bs = idx / NH;
  const int b = bs / S_LEN, s = bs % S_LEN;
  const float4 t4 = ((const float4*)htil)[((size_t)b * NH + h) * S_LEN + s];
  float t[4] = {t4.x, t4.y, t4.z, t4.w};
  const float mean = (t[0] + t[1] + t[2] + t[3]) * 0.25f;
  float var = 0.f;
#pragma unroll
  for (int d = 0; d < 4; ++d) { const float u = t[d] - mean; var += u * u; }
  const float inv = rsqrtf(var * 0.25f + 1e-5f);
#pragma unroll
  for (int d = 0; d < 4; ++d) {
    const int ci = h * 4 + d;
    const float hn = (t[d] - mean) * inv * on_g[ci] + on_b[ci];
    const float hs = hn + skip[ci] * xconv[(size_t)bs * INNER + ci];
    const float z = xi[(size_t)bs * UP_N + INNER + ci];
    const float sz = z / (1.f + __expf(-z));
    hstate_bf[(size_t)bs * INNER + ci] = (__bf16)(hs * sz);
  }
}

// ---------------- K8: down-projection GEMM (WMMA) + residual ----------------
__global__ __launch_bounds__(32) void k_down_gemm(const __bf16* __restrict__ A,   // (1152,192)
                                                  const __bf16* __restrict__ W,   // (96,192)
                                                  const float* __restrict__ bias,
                                                  const float* __restrict__ xin,  // (B,C,S)
                                                  float* __restrict__ out) {      // (B,C,H,W)
  const int nt = blockIdx.x % 6, mt = blockIdx.x / 6;
  const int m0 = mt * 16, n0 = nt * 16;
  const int lane = threadIdx.x;
  const int n = n0 + (lane & 15);
  const __bf16* wrow = W + (size_t)n * INNER;
  v8f c = {};
#pragma unroll
  for (int k0 = 0; k0 < INNER; k0 += 32) {
    v16bf a = load_a_frag_bf(A, INNER, m0, k0, lane);
    v16bf b = load_b_frag_bf(wrow, k0, lane);
    c = __builtin_amdgcn_wmma_f32_16x16x32_bf16(false, a, false, b, (short)0, c, false, false);
  }
  const int half = lane >> 4;
  const float bn = bias[n];
#pragma unroll
  for (int i = 0; i < 8; ++i) {
    const int m = m0 + i + half * 8;
    const int b = m / S_LEN, s = m % S_LEN;
    const size_t o = ((size_t)b * DIM + n) * S_LEN + s;
    out[o] = xin[o] + c[i] + bn;
  }
}

// ---------------- host launcher ----------------
extern "C" void kernel_launch(void* const* d_in, const int* in_sizes, int n_in,
                              void* d_out, int out_size, void* d_ws, size_t ws_size,
                              hipStream_t stream) {
  const float* x      = (const float*)d_in[0];
  const float* ln_g   = (const float*)d_in[1];
  const float* ln_b   = (const float*)d_in[2];
  const float* w_up   = (const float*)d_in[3];
  const float* b_up   = (const float*)d_in[4];
  const float* w_q    = (const float*)d_in[5];
  const float* b_q    = (const float*)d_in[6];
  const float* w_k    = (const float*)d_in[7];
  const float* b_k    = (const float*)d_in[8];
  const float* w_v    = (const float*)d_in[9];
  const float* b_v    = (const float*)d_in[10];
  const float* w_conv = (const float*)d_in[11];
  const float* b_conv = (const float*)d_in[12];
  const float* w_ig   = (const float*)d_in[13];
  const float* b_ig   = (const float*)d_in[14];
  const float* w_fg   = (const float*)d_in[15];
  const float* b_fg   = (const float*)d_in[16];
  const float* on_g   = (const float*)d_in[17];
  const float* on_b   = (const float*)d_in[18];
  const float* skip   = (const float*)d_in[19];
  const float* w_down = (const float*)d_in[20];
  const float* b_down = (const float*)d_in[21];
  float* out = (float*)d_out;

  // workspace carve (64B aligned slabs)
  char* p = (char*)d_ws;
  auto carve = [&](size_t bytes) -> char* {
    char* r = p; p += (bytes + 63) & ~(size_t)63; return r;
  };
  float*  xi       = (float*)carve((size_t)BS * UP_N * 4);
  float*  xconv    = (float*)carve((size_t)BS * INNER * 4);
  float*  qkv      = (float*)carve((size_t)BS * QKV_F * 4);
  float*  ig       = (float*)carve((size_t)BATCH * NH * S_LEN * 4);
  float*  fg       = (float*)carve((size_t)BATCH * NH * S_LEN * 4);
  float*  htil     = (float*)carve((size_t)BATCH * NH * S_LEN * DH * 4);
  __bf16* h_bf     = (__bf16*)carve((size_t)BS * DIM * 2);
  __bf16* qkv_bf   = (__bf16*)carve((size_t)BS * QKV_F * 2);
  __bf16* hst_bf   = (__bf16*)carve((size_t)BS * INNER * 2);
  __bf16* wup_bf   = (__bf16*)carve((size_t)UP_N * DIM * 2);
  __bf16* wig_bf   = (__bf16*)carve((size_t)NH * QKV_F * 2);
  __bf16* wfg_bf   = (__bf16*)carve((size_t)NH * QKV_F * 2);
  __bf16* wdown_bf = (__bf16*)carve((size_t)DIM * INNER * 2);

  // stage weights once per call (deterministic)
  k_cvt_bf16<<<(UP_N * DIM + 255) / 256, 256, 0, stream>>>(w_up, wup_bf, UP_N * DIM);
  k_cvt_bf16<<<(NH * QKV_F + 255) / 256, 256, 0, stream>>>(w_ig, wig_bf, NH * QKV_F);
  k_cvt_bf16<<<(NH * QKV_F + 255) / 256, 256, 0, stream>>>(w_fg, wfg_bf, NH * QKV_F);
  k_cvt_bf16<<<(DIM * INNER + 255) / 256, 256, 0, stream>>>(w_down, wdown_bf, DIM * INNER);

  k_layernorm<<<BS, 32, 0, stream>>>(x, ln_g, ln_b, h_bf);
  k_up_gemm<<<(BS / 16) * (UP_N / 16), 32, 0, stream>>>(h_bf, wup_bf, b_up, xi);
  k_conv_silu<<<(BS * INNER + 255) / 256, 256, 0, stream>>>(xi, w_conv, b_conv, xconv);
  k_qkv<<<(BS * INNER + 255) / 256, 256, 0, stream>>>(xi, xconv, w_q, b_q, w_k, b_k,
                                                      w_v, b_v, qkv, qkv_bf);
  k_gate_gemm<<<(BS / 16) * 6, 32, 0, stream>>>(qkv_bf, wig_bf, b_ig, wfg_bf, b_fg, ig, fg);
  k_mlstm<<<BATCH * NH, 256, 0, stream>>>(qkv, ig, fg, htil);
  k_fuse<<<(BS * NH + 255) / 256, 256, 0, stream>>>(htil, on_g, on_b, skip, xconv, xi, hst_bf);
  k_down_gemm<<<(BS / 16) * 6, 32, 0, stream>>>(hst_bf, wdown_bf, b_down, x, out);
}
